// RMAttention_63496796504499
// MI455X (gfx1250) — matrix operand order repaired and checked by reading, not compiled, not run on hardware
//
#include <hip/hip_runtime.h>
#include <hip/hip_bf16.h>
#include <math.h>

// ---------------- problem constants ----------------
#define BB   4
#define CC   96
#define DD   192
#define NNs  16
#define RRk  6
#define KKd  4
#define HHh  64
#define WWw  64
#define LLn  (HHh*WWw)          // 4096
#define MMr  (BB*LLn)           // 16384 rows (B*L)
#define MLPD 384
#define XPN  (RRk + 2*NNs)      // 38
#define LNEPS 1e-5f

typedef __attribute__((ext_vector_type(16))) __bf16 v16bf;
typedef __attribute__((ext_vector_type(8)))  float  v8f;

// ---------------- small device helpers ----------------
__device__ __forceinline__ unsigned short f2bf(float f) {
  unsigned u = __float_as_uint(f);
  u += 0x7fffu + ((u >> 16) & 1u);        // round-to-nearest-even
  return (unsigned short)(u >> 16);
}
__device__ __forceinline__ float gelu_f(float x) {
  float x3 = x * x * x;
  return 0.5f * x * (1.f + tanhf(0.79788456080286535588f * (x + 0.044715f * x3)));
}
__device__ __forceinline__ float silu_f(float x) {
  return x / (1.f + __expf(-x));
}

// Fragment loaders per CDNA5 ISA VGPR layouts (16-bit A 16x32, B 32x16).
__device__ __forceinline__ v16bf load_frag_A(const unsigned short (*Ash)[40], int row0, int lane) {
  union { v16bf v; uint4 q[2]; } u;
  int r = row0 + (lane & 15);
  int half = (lane >> 4) << 3;            // k-offset 0 or 8
  u.q[0] = *(const uint4 *)&Ash[r][half];
  u.q[1] = *(const uint4 *)&Ash[r][16 + half];
  return u.v;
}
__device__ __forceinline__ v16bf load_frag_B(const unsigned short (*Bsh)[40], int col0, int lane) {
  union { v16bf v; uint4 q[2]; } u;
  int c = col0 + (lane & 15);
  int ks = (lane >> 4) << 3;              // k-offset 0 or 8
  u.q[0] = *(const uint4 *)&Bsh[c][ks];
  u.q[1] = *(const uint4 *)&Bsh[c][16 + ks];
  return u.v;
}

// ---------------- generic bf16-WMMA GEMM ----------------
// C[M,N] = act(A[M,K] * B[K,N] + bias) (+BN affine) (+residual) (*rowscale)
// IM2COL: A is a virtual im2col view of NHWC image X (3x3, pad 1).
// Invariants exploited: M % 128 == 0, K % 32 == 0, all A row strides % 4 == 0,
// N % 4 == 0 with 16B-aligned base pointers.
template <int ACT, bool IM2COL>
__global__ __launch_bounds__(256) void gemm_kernel(
    const float *__restrict__ A, int lda, const float *__restrict__ Bm,
    float *__restrict__ Cm, int ldo, int M, int N, int K,
    const float *__restrict__ bias, const float *__restrict__ residual, int ldres,
    const float *__restrict__ rowscale, const float *__restrict__ colscale,
    const float *__restrict__ colshift, int Hc, int Wc, int Cin) {
  __shared__ __align__(16) unsigned short Ash[128][40];
  __shared__ __align__(16) unsigned short Bsh[64][40];
  const int tid = threadIdx.x;
  const int lane = tid & 31;
  const int wid = tid >> 5;
  const int wm = wid >> 1;                 // 0..3 (32-row band)
  const int wn = wid & 1;                  // 0..1 (32-col band)
  const int m0 = blockIdx.x * 128;
  const int n0 = blockIdx.y * 64;

  v8f acc[2][2] = {};

  for (int k0 = 0; k0 < K; k0 += 32) {
    // ---- stage A tile (128x32) as bf16 ----
    if (!IM2COL) {
      // 1024 float4s, 4 per thread; no bounds checks (exact tiling).
#pragma unroll
      for (int i = 0; i < 4; ++i) {
        int v4 = tid + i * 256;            // float4 linear index
        int mi = v4 >> 3;                  // (v4*4)/32
        int ki = (v4 & 7) << 2;
        const float4 f = *(const float4 *)(A + (size_t)(m0 + mi) * lda + (k0 + ki));
        uint2 p;
        p.x = (unsigned)f2bf(f.x) | ((unsigned)f2bf(f.y) << 16);
        p.y = (unsigned)f2bf(f.z) | ((unsigned)f2bf(f.w) << 16);
        *(uint2 *)&Ash[mi][ki] = p;
      }
    } else {
#pragma unroll
      for (int i = 0; i < 16; ++i) {
        int e = tid + i * 256;
        int mi = e >> 5, ki = e & 31;
        int gm = m0 + mi, gk = k0 + ki;
        int ci = gk % Cin;
        int rest = gk / Cin;
        int kw = rest % 3 - 1;
        int kh = rest / 3 - 1;
        int bimg = gm / (Hc * Wc);
        int l = gm - bimg * (Hc * Wc);
        int hh = l / Wc + kh;
        int ww = l % Wc + kw;
        float v = 0.f;
        if (hh >= 0 && hh < Hc && ww >= 0 && ww < Wc)
          v = A[((size_t)(bimg * Hc + hh) * Wc + ww) * Cin + ci];
        Ash[mi][ki] = f2bf(v);
      }
    }
    // ---- stage B tile (32x64) transposed -> Bsh[n][k] ----
    // 512 float4s along n, 2 per thread; single guard per vector (N tail only).
#pragma unroll
    for (int i = 0; i < 2; ++i) {
      int v4 = tid + i * 256;
      int ki = v4 >> 4;
      int ni = (v4 & 15) << 2;
      int gk = k0 + ki, gn = n0 + ni;
      float4 f = make_float4(0.f, 0.f, 0.f, 0.f);
      if (gn + 3 < N) {
        f = *(const float4 *)(Bm + (size_t)gk * N + gn);
      } else if (gn < N) {
        const float *bp = Bm + (size_t)gk * N;
        f.x = bp[gn];
        if (gn + 1 < N) f.y = bp[gn + 1];
        if (gn + 2 < N) f.z = bp[gn + 2];
      }
      Bsh[ni][ki]     = f2bf(f.x);
      Bsh[ni + 1][ki] = f2bf(f.y);
      Bsh[ni + 2][ki] = f2bf(f.z);
      Bsh[ni + 3][ki] = f2bf(f.w);
    }
    __syncthreads();

    v16bf a0 = load_frag_A(Ash, wm * 32, lane);
    v16bf a1 = load_frag_A(Ash, wm * 32 + 16, lane);
    v16bf b0 = load_frag_B(Bsh, wn * 32, lane);
    v16bf b1 = load_frag_B(Bsh, wn * 32 + 16, lane);
#if defined(__gfx1250__)
    acc[0][0] = __builtin_amdgcn_wmma_f32_16x16x32_bf16(false, a0, false, b0, (short)0, acc[0][0], false, false);
    acc[0][1] = __builtin_amdgcn_wmma_f32_16x16x32_bf16(false, a0, false, b1, (short)0, acc[0][1], false, false);
    acc[1][0] = __builtin_amdgcn_wmma_f32_16x16x32_bf16(false, a1, false, b0, (short)0, acc[1][0], false, false);
    acc[1][1] = __builtin_amdgcn_wmma_f32_16x16x32_bf16(false, a1, false, b1, (short)0, acc[1][1], false, false);
#else
    (void)a0; (void)a1; (void)b0; (void)b1;
#endif
    __syncthreads();
  }

  // epilogue: C/D layout -> m = r + 8*(lane>=16), n = lane&15
#pragma unroll
  for (int ti = 0; ti < 2; ++ti) {
#pragma unroll
    for (int tj = 0; tj < 2; ++tj) {
      int mb = m0 + wm * 32 + ti * 16 + ((lane >> 4) << 3);
      int nn = n0 + wn * 32 + tj * 16 + (lane & 15);
      if (nn >= N) continue;
      float cs = colscale ? colscale[nn] : 1.f;
      float csh = colshift ? colshift[nn] : 0.f;
      float bv = bias ? bias[nn] : 0.f;
#pragma unroll
      for (int r = 0; r < 8; ++r) {
        int m = mb + r;
        float v = acc[ti][tj][r] + bv;
        if (colscale) v = v * cs + csh;
        if (ACT == 1) v = gelu_f(v);
        if (residual) v += residual[(size_t)m * ldres + nn];
        if (rowscale) v *= rowscale[m];
        Cm[(size_t)m * ldo + nn] = v;
      }
    }
  }
}

// ---------------- LayerNorm (wave per row), optional SiLU gate ----------------
__global__ __launch_bounds__(256) void ln_kernel(
    const float *__restrict__ in, float *__restrict__ out,
    const float *__restrict__ g, const float *__restrict__ b, int rows, int C,
    const float *__restrict__ gate, int gld, int goff) {
  int row = blockIdx.x * 8 + (threadIdx.x >> 5);
  if (row >= rows) return;
  int lane = threadIdx.x & 31;
  int per = C >> 5;                       // 3 (C=96) or 6 (C=192)
  const float *rp = in + (size_t)row * C;
  float vals[6];
  float s = 0.f;
  for (int i = 0; i < per; ++i) { vals[i] = rp[i * 32 + lane]; s += vals[i]; }
  for (int o = 16; o > 0; o >>= 1) s += __shfl_xor(s, o, 32);
  float mu = s / (float)C;
  float v2 = 0.f;
  for (int i = 0; i < per; ++i) { float d = vals[i] - mu; v2 += d * d; }
  for (int o = 16; o > 0; o >>= 1) v2 += __shfl_xor(v2, o, 32);
  float inv = rsqrtf(v2 / (float)C + LNEPS);
  float *op = out + (size_t)row * C;
  for (int i = 0; i < per; ++i) {
    int c = i * 32 + lane;
    float o2 = (vals[i] - mu) * inv * g[c] + b[c];
    if (gate) {
      float z = gate[(size_t)row * gld + goff + c];
      o2 *= silu_f(z);
    }
    op[c] = o2;
  }
}

// ---------------- depthwise 3x3 conv + bias + SiLU ----------------
__global__ void dwconv_kernel(const float *__restrict__ xz, const float *__restrict__ cw,
                              const float *__restrict__ cb, float *__restrict__ xcs) {
  int idx = blockIdx.x * blockDim.x + threadIdx.x;
  if (idx >= MMr * DD) return;
  int d = idx % DD;
  int bl = idx / DD;
  int b = bl / LLn;
  int l = bl - b * LLn;
  int h = l / WWw, w = l % WWw;
  float acc = cb[d];
#pragma unroll
  for (int kh = -1; kh <= 1; ++kh) {
    int hh = h + kh;
    if (hh < 0 || hh >= HHh) continue;
#pragma unroll
    for (int kw = -1; kw <= 1; ++kw) {
      int ww = w + kw;
      if (ww < 0 || ww >= WWw) continue;
      acc += xz[(size_t)(b * LLn + hh * WWw + ww) * (2 * DD) + d] *
             cw[d * 9 + (kh + 1) * 3 + (kw + 1)];
    }
  }
  xcs[idx] = silu_f(acc);
}

// ---------------- selective scan (thread = (d,n) state) ----------------
// grid: BB*KKd*(DD/16) blocks of 256; writes y at SPATIAL hw-index directly,
// handling the 4 cross-scan directions via traversal order + index mapping.
__global__ __launch_bounds__(256) void scan_kernel(
    const float *__restrict__ dbl, const float *__restrict__ xcs,
    const float *__restrict__ dtw, const float *__restrict__ dtb,
    const float *__restrict__ Alog, float *__restrict__ ydir) {
  const int nblkd = DD / 16;              // 12
  int blk = blockIdx.x;
  int dblk = blk % nblkd;
  int dir = (blk / nblkd) & 3;
  int b = blk / (nblkd * KKd);
  int tid = threadIdx.x;
  int n = tid & 15;
  int d = dblk * 16 + (tid >> 4);

  float A = -__expf(Alog[(size_t)(dir * DD + d) * NNs + n]);
  float w[RRk];
#pragma unroll
  for (int r = 0; r < RRk; ++r) w[r] = dtw[(size_t)(dir * DD + d) * RRk + r];
  float bdt = dtb[dir * DD + d];

  const float *dblB = dbl + (size_t)b * LLn * (KKd * XPN) + dir * XPN;
  const float *uB = xcs + (size_t)b * LLn * DD + d;
  float *yB = ydir + ((size_t)(dir * BB + b) * LLn) * DD + d;
  const bool rev = (dir >= 2);
  const bool wh = (dir & 1);

  float h = 0.f;
  for (int t = 0; t < LLn; ++t) {
    int p = rev ? (LLn - 1 - t) : t;
    int lsp;
    if (wh) { int wq = p / HHh; int hq = p - wq * HHh; lsp = hq * WWw + wq; }
    else lsp = p;

    const float *db = dblB + (size_t)lsp * (KKd * XPN);
    float dt = bdt;
#pragma unroll
    for (int r = 0; r < RRk; ++r) dt += w[r] * db[r];
    float delta = (dt > 20.f) ? dt : log1pf(__expf(dt));
    float u = uB[(size_t)lsp * DD];
    float Bv = db[RRk + n];
    float Cv = db[RRk + NNs + n];
    h = h * __expf(delta * A) + (delta * u) * Bv;
    float part = h * Cv;
    part += __shfl_xor(part, 1, 32);
    part += __shfl_xor(part, 2, 32);
    part += __shfl_xor(part, 4, 32);
    part += __shfl_xor(part, 8, 32);
    if (n == 0) yB[(size_t)lsp * DD] = part;

    if (t + 1 < LLn) {                    // prefetch next step's rows
      int pn = rev ? (LLn - 2 - t) : t + 1;
      int lspn;
      if (wh) { int wq = pn / HHh; int hq = pn - wq * HHh; lspn = hq * WWw + wq; }
      else lspn = pn;
      __builtin_prefetch(dblB + (size_t)lspn * (KKd * XPN), 0, 0);
      __builtin_prefetch(uB + (size_t)lspn * DD, 0, 0);
    }
  }
}

// ---------------- combine 4 directions + D*u skip ----------------
__global__ void combine_kernel(const float *__restrict__ ydir, const float *__restrict__ xcs,
                               const float *__restrict__ Ds, float *__restrict__ y) {
  int idx = blockIdx.x * blockDim.x + threadIdx.x;
  if (idx >= MMr * DD) return;
  int d = idx % DD;
  size_t per = (size_t)BB * LLn * DD;
  float s = ydir[idx] + ydir[per + idx] + ydir[2 * per + idx] + ydir[3 * per + idx];
  float ds = Ds[d] + Ds[DD + d] + Ds[2 * DD + d] + Ds[3 * DD + d];
  y[idx] = s + xcs[idx] * ds;
}

// ---------------- layout / pack helpers ----------------
__global__ void nchw2nhwc_kernel(const float *__restrict__ x, float *__restrict__ o) {
  int idx = blockIdx.x * blockDim.x + threadIdx.x;
  if (idx >= MMr * CC) return;
  int c = idx % CC;
  int bl = idx / CC;
  int b = bl / LLn;
  int l = bl - b * LLn;
  o[idx] = x[((size_t)b * CC + c) * LLn + l];
}
__global__ void add_kernel(const float *__restrict__ a, const float *__restrict__ b,
                           float *__restrict__ o, int n) {
  int idx = blockIdx.x * blockDim.x + threadIdx.x;
  if (idx < n) o[idx] = a[idx] + b[idx];
}
__global__ void pack_xproj_kernel(const float *__restrict__ xw, float *__restrict__ wx) {
  int idx = blockIdx.x * blockDim.x + threadIdx.x;
  if (idx >= DD * KKd * XPN) return;
  int d = idx / (KKd * XPN);
  int j = idx - d * (KKd * XPN);          // j = dir*38 + c
  wx[idx] = xw[(size_t)j * DD + d];       // x_proj_w[dir,c,d]
}
__global__ void pack_conv_kernel(const float *__restrict__ w, float *__restrict__ wpk) {
  int idx = blockIdx.x * blockDim.x + threadIdx.x;
  if (idx >= 9 * CC * CC) return;
  int co = idx % CC;
  int kidx = idx / CC;
  int ci = kidx % CC;
  int r = kidx / CC;                      // kh*3+kw
  wpk[idx] = w[((size_t)(co * CC + ci)) * 9 + r];
}
__global__ void bnpack_kernel(const float *__restrict__ g, const float *__restrict__ b,
                              const float *__restrict__ m, const float *__restrict__ v,
                              float *__restrict__ sc, float *__restrict__ sh) {
  int c = threadIdx.x;
  if (c < CC) {
    float s = g[c] * rsqrtf(v[c] + LNEPS);
    sc[c] = s;
    sh[c] = b[c] - m[c] * s;
  }
}

// ---------------- bilinear resize 32->64 + attention mask ----------------
__global__ void resize_kernel(const float *__restrict__ mp, float *__restrict__ mpf,
                              float *__restrict__ attn) {
  int idx = blockIdx.x * blockDim.x + threadIdx.x;
  if (idx >= BB * LLn) return;
  int b = idx / LLn;
  int l = idx - b * LLn;
  int oh = l / WWw, ow = l % WWw;
  float sy = oh * 0.5f - 0.25f;
  float sx = ow * 0.5f - 0.25f;
  int y0 = (int)floorf(sy), x0 = (int)floorf(sx);
  float fy = sy - (float)y0, fx = sx - (float)x0;
  int y0c = min(max(y0, 0), 31), y1c = min(max(y0 + 1, 0), 31);
  int x0c = min(max(x0, 0), 31), x1c = min(max(x0 + 1, 0), 31);
  const float *mb = mp + (size_t)b * 1024;
  float v = (1.f - fy) * ((1.f - fx) * mb[y0c * 32 + x0c] + fx * mb[y0c * 32 + x1c]) +
            fy * ((1.f - fx) * mb[y1c * 32 + x0c] + fx * mb[y1c * 32 + x1c]);
  mpf[idx] = v;
  attn[idx] = 1.f / (1.f + __expf(v));    // 1 - sigmoid(v)
}

// ---------------- final 1x1 conv + bias + mp add ----------------
__global__ void final_kernel(const float *__restrict__ y2, const float *__restrict__ w3,
                             const float *__restrict__ b3, const float *__restrict__ mpf,
                             float *__restrict__ out) {
  int idx = blockIdx.x * blockDim.x + threadIdx.x;
  if (idx >= BB * LLn) return;
  const float *rp = y2 + (size_t)idx * CC;
  float s = b3[0] + mpf[idx];
  for (int c = 0; c < CC; ++c) s += rp[c] * w3[c];
  out[idx] = s;
}

// ---------------- host orchestration ----------------
extern "C" void kernel_launch(void *const *d_in, const int *in_sizes, int n_in,
                              void *d_out, int out_size, void *d_ws, size_t ws_size,
                              hipStream_t stream) {
  (void)in_sizes; (void)n_in; (void)out_size; (void)ws_size;
  auto P = [&](int i) { return (const float *)d_in[i]; };
  const float *x = P(0);
  const float *mapv = P(1);

  size_t off = 0;
  auto alloc = [&](size_t nfloats) {
    float *p = (float *)((char *)d_ws + off);
    off += nfloats * sizeof(float);
    return p;
  };
  float *xbuf = alloc((size_t)MMr * CC);      // input NHWC
  float *t    = alloc((size_t)MMr * CC);      // LN scratch
  float *xz   = alloc((size_t)MMr * 2 * DD);  // in_proj out (xc | z)
  float *xcs  = alloc((size_t)MMr * DD);      // conv+silu
  float *dbl  = alloc((size_t)MMr * KKd * XPN);
  float *ydir = alloc((size_t)KKd * MMr * DD);
  float *ybuf = alloc((size_t)MMr * DD);
  float *x1   = alloc((size_t)MMr * CC);
  float *hbuf = alloc((size_t)MMr * MLPD);
  float *vout = alloc((size_t)MMr * CC);
  float *bufA = alloc((size_t)MMr * CC);
  float *sum2 = alloc((size_t)MMr * CC);
  float *y2   = alloc((size_t)MMr * CC);
  float *mpf  = alloc((size_t)BB * LLn);
  float *attn = alloc((size_t)BB * LLn);
  float *wx   = alloc((size_t)DD * KKd * XPN);
  float *wpk  = alloc((size_t)9 * CC * CC);
  float *bnsc = alloc(CC);
  float *bnsh = alloc(CC);

  const float *NF = nullptr;
  dim3 gN96(MMr / 128, (CC + 63) / 64);

  // full VSS block: in -> vout2 (param base pb; optional per-row attn scale)
  auto vss = [&](int pb, const float *vin, float *vout2, const float *rowscale) {
    ln_kernel<<<MMr / 8, 256, 0, stream>>>(vin, t, P(pb + 0), P(pb + 1), MMr, CC, NF, 0, 0);
    dim3 g1(MMr / 128, (2 * DD + 63) / 64);
    gemm_kernel<0, false><<<g1, 256, 0, stream>>>(t, CC, P(pb + 2), xz, 2 * DD, MMr, 2 * DD, CC,
                                                  NF, NF, 0, NF, NF, NF, 0, 0, 0);
    dwconv_kernel<<<(MMr * DD + 255) / 256, 256, 0, stream>>>(xz, P(pb + 3), P(pb + 4), xcs);
    pack_xproj_kernel<<<(DD * KKd * XPN + 255) / 256, 256, 0, stream>>>(P(pb + 5), wx);
    dim3 g2(MMr / 128, (KKd * XPN + 63) / 64);
    gemm_kernel<0, false><<<g2, 256, 0, stream>>>(xcs, DD, wx, dbl, KKd * XPN, MMr, KKd * XPN, DD,
                                                  NF, NF, 0, NF, NF, NF, 0, 0, 0);
    scan_kernel<<<BB * KKd * (DD / 16), 256, 0, stream>>>(dbl, xcs, P(pb + 6), P(pb + 7),
                                                          P(pb + 8), ydir);
    combine_kernel<<<(MMr * DD + 255) / 256, 256, 0, stream>>>(ydir, xcs, P(pb + 9), ybuf);
    // out_norm LN (in place) fused with SiLU(z) gate
    ln_kernel<<<MMr / 8, 256, 0, stream>>>(ybuf, ybuf, P(pb + 10), P(pb + 11), MMr, DD,
                                           xz, 2 * DD, DD);
    // out_proj + residual(x)
    gemm_kernel<0, false><<<gN96, 256, 0, stream>>>(ybuf, DD, P(pb + 12), x1, CC, MMr, CC, DD,
                                                    NF, vin, CC, NF, NF, NF, 0, 0, 0);
    // MLP
    ln_kernel<<<MMr / 8, 256, 0, stream>>>(x1, t, P(pb + 13), P(pb + 14), MMr, CC, NF, 0, 0);
    dim3 g4(MMr / 128, (MLPD + 63) / 64);
    gemm_kernel<1, false><<<g4, 256, 0, stream>>>(t, CC, P(pb + 15), hbuf, MLPD, MMr, MLPD, CC,
                                                  P(pb + 16), NF, 0, NF, NF, NF, 0, 0, 0);
    gemm_kernel<0, false><<<gN96, 256, 0, stream>>>(hbuf, MLPD, P(pb + 17), vout2, CC, MMr, CC,
                                                    MLPD, P(pb + 18), x1, CC, rowscale, NF, NF,
                                                    0, 0, 0);
  };

  auto conv_bn = [&](const float *cin, const float *wraw, const float *bg, const float *bb,
                     const float *bm, const float *bv, float *cout) {
    pack_conv_kernel<<<(9 * CC * CC + 255) / 256, 256, 0, stream>>>(wraw, wpk);
    bnpack_kernel<<<1, 128, 0, stream>>>(bg, bb, bm, bv, bnsc, bnsh);
    gemm_kernel<0, true><<<gN96, 256, 0, stream>>>(cin, CC, wpk, cout, CC, MMr, CC, 9 * CC,
                                                   NF, NF, 0, NF, bnsc, bnsh, HHh, WWw, CC);
  };

  // pipeline
  nchw2nhwc_kernel<<<(MMr * CC + 255) / 256, 256, 0, stream>>>(x, xbuf);
  resize_kernel<<<(BB * LLn + 255) / 256, 256, 0, stream>>>(mapv, mpf, attn);

  vss(2, xbuf, vout, attn);                               // block b1, fused attn*y
  conv_bn(vout, P(40), P(41), P(42), P(43), P(44), bufA); // conv1 + bn1
  vss(21, bufA, vout, nullptr);                           // block b2
  add_kernel<<<(MMr * CC + 255) / 256, 256, 0, stream>>>(vout, xbuf, sum2, MMr * CC);
  conv_bn(sum2, P(45), P(46), P(47), P(48), P(49), y2);   // conv2 + bn2
  final_kernel<<<(BB * LLn + 255) / 256, 256, 0, stream>>>(y2, P(50), P(51), mpf,
                                                           (float *)d_out);
}
